// RNNchain_3410204033765
// MI455X (gfx1250) — matrix operand order repaired and checked by reading, not compiled, not run on hardware
//
#include <hip/hip_runtime.h>

#define B_ 2048
#define T_ 512
#define H_ 256
#define V_ 64

typedef __attribute__((ext_vector_type(16))) __bf16 v16bf;
typedef __attribute__((ext_vector_type(8)))  __bf16 v8bf;
typedef __attribute__((ext_vector_type(8)))  float  v8f;

__device__ __forceinline__ unsigned short f2bf(float f) {
    union { float f; unsigned int u; } c; c.f = f;
    // round-to-nearest-even bf16 truncation
    return (unsigned short)((c.u + 0x7FFFu + ((c.u >> 16) & 1u)) >> 16);
}

// ---------------------------------------------------------------------------
// Kernel 1: projE[v][o] = b[o] + sum_h emb[v][h] * Wx[o][h],  Wx = W[:, :H]
// 64x256 outputs (4.2M MACs) -- replaces the full 137-GFLOP embedding GEMM.
// ---------------------------------------------------------------------------
__global__ void rnn_proj_table(const float* __restrict__ emb,
                               const float* __restrict__ W,
                               const float* __restrict__ bias,
                               float* __restrict__ projE) {
    const int v = blockIdx.x;    // 0..63
    const int o = threadIdx.x;   // 0..255
    const float* er = emb + v * H_;
    const float* wr = W + o * (2 * H_);   // row o of W, first H cols = Wx[o][:]
    float s = bias[o];
#pragma unroll 8
    for (int h = 0; h < H_; ++h) s += er[h] * wr[h];
    projE[v * H_ + o] = s;
}

// ---------------------------------------------------------------------------
// Kernel 2: the recurrence. One block = 16 batch rows through all T steps.
// 512 threads = 16 waves; wave w owns output columns [16w, 16w+16).
// Per step: all LDS loads (A operands + projE gathers) issue before the
// barrier (where s_wait_dscnt 0 is mandatory anyway), then 8 back-to-back
// v_wmma_f32_16x16x32_bf16 in two accumulator chains, then a short tail.
// sched_barrier(0) stops the scheduler from hoisting WMMAs into the loads.
// ---------------------------------------------------------------------------
__global__ void __launch_bounds__(512)
rnn_chain(const int* __restrict__ tokens,
          const float* __restrict__ h0,
          const float* __restrict__ W,
          const float* __restrict__ projE,
          float* __restrict__ out) {
    extern __shared__ char smem[];
    float*          projL = (float*)smem;                               // 64 KB
    unsigned short* WhL   = (unsigned short*)(smem + 65536);            // 128 KB (bf16)
    unsigned short* hL    = (unsigned short*)(smem + 65536 + 131072);   // 8 KB (bf16)
    int*            tokL  = (int*)(smem + 65536 + 131072 + 8192);       // 2x16 ints

    const int tid   = threadIdx.x;
    const int lane  = tid & 31;
    const int wave  = tid >> 5;            // 0..15 -> N tile
    const int bbase = blockIdx.x * 16;     // batch rows handled by this block

    // ---- prologue: stage tables into LDS --------------------------------
    for (int i = tid; i < V_ * H_; i += 512) projL[i] = projE[i];
    for (int i = tid; i < H_ * H_; i += 512) {
        const int o = i >> 8, k = i & 255;
        WhL[i] = f2bf(W[o * (2 * H_) + H_ + k]);   // Wh[o][k] = W[o][H+k]
    }
    for (int i = tid; i < 16 * H_; i += 512) {
        const int m = i >> 8, n = i & 255;
        hL[i] = f2bf(h0[(bbase + m) * H_ + n]);
    }
    if (tid < 16) tokL[tid] = tokens[(bbase + tid) * T_ + 0];  // buffer 0, t=0
    __syncthreads();

    // ---- preload B operand (Wh strip for this wave's N tile), once ------
    // 16-bit B layout: lane L holds column N=L%16; lanes 0-15: K=c*32+{0..7,16..23},
    // lanes 16-31: same +8, packed pairs across the 8 VGPRs.
    const int n0 = wave * 16;
    const int nB = n0 + (lane & 15);
    const int kB = (lane & 16) >> 1;       // 0 or 8
    v16bf breg[8];
#pragma unroll
    for (int c = 0; c < 8; ++c) {
        const unsigned short* p = &WhL[nB * H_ + c * 32 + kB];
        v8bf lo = *reinterpret_cast<const v8bf*>(p);
        v8bf hi = *reinterpret_cast<const v8bf*>(p + 16);
        breg[c] = __builtin_shufflevector(lo, hi,
                   0,1,2,3,4,5,6,7,8,9,10,11,12,13,14,15);
    }

    // 16-bit A layout: lane L holds row M=L%16; lanes 0-15: K=c*32+{0..7,16..23},
    // lanes 16-31: same +8.
    const int mA = lane & 15;
    const int kA = (lane & 16) >> 1;       // 0 or 8
    // D layout: VGPR r -> row (r + 8*(lane>=16)), column n0 + lane%16
    const int Mb = (lane & 16) >> 1;       // 0 or 8
    const int nD = n0 + (lane & 15);

    for (int t = 0; t < T_; ++t) {
        __syncthreads();                   // barrier 1: hL from prev step stable

        // Prefetch NEXT step's tokens into the other tokL buffer (global,
        // L2-resident: token rows are contiguous along t).
        if (tid < 16 && t + 1 < T_)
            tokL[((t + 1) & 1) * 16 + tid] = tokens[(bbase + tid) * T_ + t + 1];

        // This step's tokens + projE gather -- independent of h, issued
        // concurrently with the A loads so the post-WMMA tail is short.
        const int* tkb = &tokL[(t & 1) * 16 + Mb];
        int tk[8];
        {
            const int4 q0 = *reinterpret_cast<const int4*>(tkb);
            const int4 q1 = *reinterpret_cast<const int4*>(tkb + 4);
            tk[0]=q0.x; tk[1]=q0.y; tk[2]=q0.z; tk[3]=q0.w;
            tk[4]=q1.x; tk[5]=q1.y; tk[6]=q1.z; tk[7]=q1.w;
        }
        float pv[8];
#pragma unroll
        for (int r = 0; r < 8; ++r) pv[r] = projL[tk[r] * H_ + nD];

        // A operand loads: 16 x ds_load_b128 into distinct registers.
        v8bf alo[8], ahi[8];
#pragma unroll
        for (int c = 0; c < 8; ++c) {
            const unsigned short* p = &hL[mA * H_ + c * 32 + kA];
            alo[c] = *reinterpret_cast<const v8bf*>(p);
            ahi[c] = *reinterpret_cast<const v8bf*>(p + 16);
        }
        __syncthreads();                   // barrier 2 (s_wait_dscnt 0 here)
        __builtin_amdgcn_sched_barrier(0); // keep WMMAs below, loads above

        // Two independent accumulator chains (even/odd K chunks) for ILP.
        v8f acc0 = {}, acc1 = {};
#pragma unroll
        for (int c = 0; c < 8; c += 2) {
            v16bf a0 = __builtin_shufflevector(alo[c], ahi[c],
                        0,1,2,3,4,5,6,7,8,9,10,11,12,13,14,15);
            acc0 = __builtin_amdgcn_wmma_f32_16x16x32_bf16(
                       false, a0, false, breg[c], (short)0, acc0, false, false);
            v16bf a1 = __builtin_shufflevector(alo[c + 1], ahi[c + 1],
                        0,1,2,3,4,5,6,7,8,9,10,11,12,13,14,15);
            acc1 = __builtin_amdgcn_wmma_f32_16x16x32_bf16(
                       false, a1, false, breg[c + 1], (short)0, acc1, false, false);
        }
        v8f acc = acc0 + acc1;

        // Tail: sigmoid + stores only (proj values already in registers).
        float* orow = out + (size_t)t * (size_t)(B_ * H_)
                          + (size_t)(bbase + Mb) * H_ + nD;
#pragma unroll
        for (int r = 0; r < 8; ++r) {
            const float x = pv[r] + acc[r];
            const float s = __builtin_amdgcn_rcpf(1.0f + __expf(-x));
            orow[(size_t)r * H_] = s;                  // hs[t][bbase+Mb+r][nD]
            hL[(Mb + r) * H_ + nD] = f2bf(s);          // next step's A operand
        }
    }
}

// ---------------------------------------------------------------------------
extern "C" void kernel_launch(void* const* d_in, const int* in_sizes, int n_in,
                              void* d_out, int out_size, void* d_ws, size_t ws_size,
                              hipStream_t stream) {
    const int*   tokens = (const int*)  d_in[0];
    const float* h0     = (const float*)d_in[1];
    const float* emb    = (const float*)d_in[2];
    const float* W      = (const float*)d_in[3];
    const float* bias   = (const float*)d_in[4];
    float* out   = (float*)d_out;
    float* projE = (float*)d_ws;       // 64*256 f32 = 64 KB scratch

    const size_t smem = 65536 + 131072 + 8192 + 128;  // ~205 KB < 320 KB/WGP
    (void)hipFuncSetAttribute((const void*)rnn_chain,
                              hipFuncAttributeMaxDynamicSharedMemorySize,
                              (int)smem);

    rnn_proj_table<<<V_, H_, 0, stream>>>(emb, W, bias, projE);
    rnn_chain<<<B_ / 16, 512, smem, stream>>>(tokens, h0, W, projE, out);
}
